// AnyAttention_30786325578194
// MI455X (gfx1250) — compile-verified, matrix operand order checked by reading, not run on hardware
//
#include <hip/hip_runtime.h>
#include <hip/hip_bf16.h>

// Problem constants (match reference)
constexpr int CB  = 2;
constexpr int CNQ = 2048;
constexpr int CNK = 2048;
constexpr int CD  = 1024;
constexpr int CH  = 16;
constexpr int CHEAD = 64;
constexpr float CSCALE = 0.125f;  // 1/sqrt(64)

typedef __attribute__((ext_vector_type(16))) _Float16 v16h;
typedef __attribute__((ext_vector_type(8)))  float    v8f;

// 16-byte int vector matching the async-LDS builtin's pointee type
typedef int v4i __attribute__((vector_size(4 * sizeof(int))));
typedef __attribute__((address_space(1))) v4i gv4i_t;  // global (prints as __device__)
typedef __attribute__((address_space(3))) v4i lv4i_t;  // LDS

// ---------------------------------------------------------------------------
// CDNA5 async global->LDS copy (16B per lane), with safe fallback
// ---------------------------------------------------------------------------
__device__ inline void async_copy_b128(const _Float16* __restrict__ g,
                                       _Float16* __restrict__ l) {
#if __has_builtin(__builtin_amdgcn_global_load_async_to_lds_b128)
  __builtin_amdgcn_global_load_async_to_lds_b128(
      (gv4i_t*)(uintptr_t)g, (lv4i_t*)(uint32_t)(uintptr_t)l,
      /*ioffset=*/0, /*cpol=*/0);
#else
  *(uint4*)l = *(const uint4*)g;
#endif
}

template <int N>
__device__ inline void wait_async() {
#if __has_builtin(__builtin_amdgcn_s_wait_asynccnt)
  __builtin_amdgcn_s_wait_asynccnt(N);
#else
  asm volatile("s_wait_asynccnt %0" ::"n"(N) : "memory");
#endif
}

// ---------------------------------------------------------------------------
// WMMA helpers (CDNA5 wave32, 16x16x32 f16 -> f32)
// ---------------------------------------------------------------------------
__device__ inline v8f wmma_f16(v16h a, v16h b, v8f c) {
  return __builtin_amdgcn_wmma_f32_16x16x32_f16(
      /*neg_a=*/false, a, /*neg_b=*/false, b,
      /*c_mod=*/(short)0, c, /*reuse_a=*/false, /*reuse_b=*/false);
}

// A-matrix fragment (16x32, MxK), per ISA 7.12.2:
//   lane&15 = M; VGPR v, half p -> K = (v>=4?16:0) + (lane>=16?8:0) + (v&3)*2 + p
__device__ inline v16h load_a_frag(const _Float16* __restrict__ p, int ld, int lane) {
  const _Float16* r = p + (size_t)(lane & 15) * ld;
  const int khalf = (lane >> 4) << 3;
  v16h a;
#pragma unroll
  for (int v = 0; v < 8; ++v) {
    const int kb = ((v & 4) << 2) + khalf + ((v & 3) << 1);
    a[2 * v]     = r[kb];
    a[2 * v + 1] = r[kb + 1];
  }
  return a;
}

// B-matrix fragment (32x16, KxN): lane&15 = N, same K pattern, row stride ld.
__device__ inline v16h load_b_frag(const _Float16* __restrict__ p, int ld, int lane) {
  const _Float16* c = p + (lane & 15);
  const int khalf = (lane >> 4) << 3;
  v16h b;
#pragma unroll
  for (int v = 0; v < 8; ++v) {
    const int kb = ((v & 4) << 2) + khalf + ((v & 3) << 1);
    b[2 * v]     = c[(size_t)kb * ld];
    b[2 * v + 1] = c[(size_t)(kb + 1) * ld];
  }
  return b;
}

// ---------------------------------------------------------------------------
// Fused add-pos + LayerNorm + f32->f16, one block (256 thr) per row of D=1024
// ---------------------------------------------------------------------------
__global__ __launch_bounds__(256) void ln_kernel(const float* __restrict__ x,
                                                 const float* __restrict__ pos,
                                                 const float* __restrict__ gamma,
                                                 const float* __restrict__ beta,
                                                 _Float16* __restrict__ out) {
  __shared__ float ssum[8], ssum2[8];
  const int row = blockIdx.x;
  const int t = threadIdx.x;
  const float* xr = x + (size_t)row * CD;
  float4 xv = ((const float4*)xr)[t];
  if (pos != nullptr) {
    const float4 pv = ((const float4*)(pos + (size_t)row * CD))[t];
    xv.x += pv.x; xv.y += pv.y; xv.z += pv.z; xv.w += pv.w;
  }
  float s  = xv.x + xv.y + xv.z + xv.w;
  float s2 = xv.x * xv.x + xv.y * xv.y + xv.z * xv.z + xv.w * xv.w;
#pragma unroll
  for (int off = 16; off > 0; off >>= 1) {
    s  += __shfl_xor(s,  off, 32);
    s2 += __shfl_xor(s2, off, 32);
  }
  const int lane = t & 31, wave = t >> 5;
  if (lane == 0) { ssum[wave] = s; ssum2[wave] = s2; }
  __syncthreads();
  float ts = 0.f, ts2 = 0.f;
#pragma unroll
  for (int i = 0; i < 8; ++i) { ts += ssum[i]; ts2 += ssum2[i]; }
  const float mean = ts * (1.0f / CD);
  const float var  = ts2 * (1.0f / CD) - mean * mean;
  const float rstd = rsqrtf(var + 1e-5f);
  const float xs[4] = {xv.x, xv.y, xv.z, xv.w};
#pragma unroll
  for (int j = 0; j < 4; ++j) {
    const int c = t * 4 + j;
    const float y = (xs[j] - mean) * rstd * gamma[c] + beta[c];
    out[(size_t)row * CD + c] = (_Float16)y;
  }
}

// ---------------------------------------------------------------------------
// f32 -> f16 elementwise (weights)
// ---------------------------------------------------------------------------
__global__ __launch_bounds__(256) void cvt_kernel(const float* __restrict__ src,
                                                  _Float16* __restrict__ dst, int n) {
  const int i = blockIdx.x * 256 + threadIdx.x;
  if (i < n) dst[i] = (_Float16)src[i];
}

// ---------------------------------------------------------------------------
// Tiled WMMA GEMM: out[M,N] = A[M,K] @ W[K,N] (+bias). Block tile 128x128,
// 8 waves, each wave computes 32x64 via 2x4 WMMA tiles, K-step 32.
// Double-buffered LDS, async global->LDS staging overlapped with WMMA.
// ---------------------------------------------------------------------------
__device__ inline void gemm_issue_tile(const _Float16* __restrict__ A,
                                       const _Float16* __restrict__ W,
                                       int Kdim, int Ndim, int m0, int n0, int k0,
                                       _Float16 (*As)[40], _Float16 (*Ws)[136],
                                       int t) {
  int idx = t;  // A tile: 128 rows x 32 halves = 512 x 16B
#pragma unroll
  for (int i = 0; i < 2; ++i, idx += 256) {
    const int row = idx >> 2, seg = idx & 3;
    async_copy_b128(A + (size_t)(m0 + row) * Kdim + k0 + seg * 8, &As[row][seg * 8]);
  }
  idx = t;      // W tile: 32 rows x 128 halves = 512 x 16B
#pragma unroll
  for (int i = 0; i < 2; ++i, idx += 256) {
    const int row = idx >> 4, seg = idx & 15;
    async_copy_b128(W + (size_t)(k0 + row) * Ndim + n0 + seg * 8, &Ws[row][seg * 8]);
  }
}

template <bool OUT_F32>
__global__ __launch_bounds__(256) void gemm_kernel(const _Float16* __restrict__ A,
                                                   const _Float16* __restrict__ W,
                                                   const float* __restrict__ bias,
                                                   void* __restrict__ outp,
                                                   int Kdim, int Ndim) {
  __shared__ __align__(16) _Float16 As[2][128][40];  // 32 K + 8 pad per row
  __shared__ __align__(16) _Float16 Ws[2][32][136];  // 128 N + 8 pad per row
  const int t = threadIdx.x, lane = t & 31, wave = t >> 5;
  const int m0 = blockIdx.y * 128;
  const int n0 = blockIdx.x * 128;
  const int mbase = (wave & 3) * 32;
  const int nbase = (wave >> 2) * 64;
  v8f acc[2][4];
#pragma unroll
  for (int mt = 0; mt < 2; ++mt)
#pragma unroll
    for (int nt = 0; nt < 4; ++nt) acc[mt][nt] = v8f{};

  gemm_issue_tile(A, W, Kdim, Ndim, m0, n0, 0, As[0], Ws[0], t);
  int buf = 0;
  for (int k0 = 0; k0 < Kdim; k0 += 32) {
    wait_async<0>();
    __syncthreads();  // all waves done with previous compute + this tile resident
    if (k0 + 32 < Kdim)
      gemm_issue_tile(A, W, Kdim, Ndim, m0, n0, k0 + 32, As[buf ^ 1], Ws[buf ^ 1], t);

    const v16h a0 = load_a_frag(&As[buf][mbase][0], 40, lane);
    const v16h a1 = load_a_frag(&As[buf][mbase + 16][0], 40, lane);
    v16h bf[4];
#pragma unroll
    for (int nt = 0; nt < 4; ++nt)
      bf[nt] = load_b_frag(&Ws[buf][0][nbase + nt * 16], 136, lane);
#pragma unroll
    for (int nt = 0; nt < 4; ++nt) {
      acc[0][nt] = wmma_f16(a0, bf[nt], acc[0][nt]);
      acc[1][nt] = wmma_f16(a1, bf[nt], acc[1][nt]);
    }
    buf ^= 1;
  }

#pragma unroll
  for (int mt = 0; mt < 2; ++mt) {
#pragma unroll
    for (int nt = 0; nt < 4; ++nt) {
      const int ncol = n0 + nbase + nt * 16 + (lane & 15);
      const float bv = (OUT_F32 && bias) ? bias[ncol] : 0.f;
#pragma unroll
      for (int r = 0; r < 8; ++r) {
        const int mr = m0 + mbase + mt * 16 + ((lane >> 4) << 3) + r;
        if (OUT_F32) {
          ((float*)outp)[(size_t)mr * Ndim + ncol] = acc[mt][nt][r] + bv;
        } else {
          ((_Float16*)outp)[(size_t)mr * Ndim + ncol] = (_Float16)acc[mt][nt][r];
        }
      }
    }
  }
}

// ---------------------------------------------------------------------------
// Flash attention: one wave per (b, h, 16-row Q tile), 2 waves/block.
// KV tile = 32, double-buffered async K/V staging (16 loads in flight).
// S = Q K^T via WMMA; online softmax; P via LDS relayout for P@V WMMA.
// ---------------------------------------------------------------------------
__device__ inline void attn_issue_tile(const _Float16* __restrict__ Kh,
                                       const _Float16* __restrict__ Vh,
                                       size_t krow0, int kv0, int h, int lane,
                                       _Float16 (*Ks)[CHEAD], _Float16 (*Vs)[CHEAD]) {
  const _Float16* kp = Kh + (krow0 + kv0 + lane) * CD + h * CHEAD;
  const _Float16* vp = Vh + (krow0 + kv0 + lane) * CD + h * CHEAD;
#pragma unroll
  for (int i = 0; i < 8; ++i) {
    async_copy_b128(kp + i * 8, &Ks[lane][i * 8]);
    async_copy_b128(vp + i * 8, &Vs[lane][i * 8]);
  }
}

__global__ __launch_bounds__(64) void attn_kernel(const _Float16* __restrict__ Qh,
                                                  const _Float16* __restrict__ Kh,
                                                  const _Float16* __restrict__ Vh,
                                                  _Float16* __restrict__ Oh) {
  __shared__ __align__(16) _Float16 Ks[2][2][32][CHEAD];  // [wave][buf][kv][c]
  __shared__ __align__(16) _Float16 Vs[2][2][32][CHEAD];
  __shared__ __align__(16) _Float16 Ps[2][16][32];
  const int lane = threadIdx.x & 31;
  const int wave = threadIdx.x >> 5;
  const int task = blockIdx.x * 2 + wave;         // B*H*(NQ/16) tasks
  const int qtile = task & (CNQ / 16 - 1);
  const int bh = task >> 7;                       // NQ/16 == 128
  const int h = bh & (CH - 1);
  const int b = bh >> 4;
  const size_t qrow0 = (size_t)b * CNQ + (size_t)qtile * 16;
  const size_t krow0 = (size_t)b * CNK;

  // Q fragments resident for the whole KV sweep
  const _Float16* qp = Qh + qrow0 * CD + h * CHEAD;
  const v16h qa0 = load_a_frag(qp, CD, lane);       // c 0..31
  const v16h qa1 = load_a_frag(qp + 32, CD, lane);  // c 32..63

  v8f o0 = {}, o1 = {}, o2 = {}, o3 = {};
  float mrow[8], lrow[8];
#pragma unroll
  for (int r = 0; r < 8; ++r) { mrow[r] = -3.0e38f; lrow[r] = 0.f; }

  attn_issue_tile(Kh, Vh, krow0, 0, h, lane, Ks[wave][0], Vs[wave][0]);

  for (int kv0 = 0; kv0 < CNK; kv0 += 32) {
    const int buf = (kv0 >> 5) & 1;
    if (kv0 + 32 < CNK)
      attn_issue_tile(Kh, Vh, krow0, kv0 + 32, h, lane,
                      Ks[wave][buf ^ 1], Vs[wave][buf ^ 1]);
    wait_async<16>();  // retire current tile; next tile (16 loads) stays in flight
    asm volatile("s_wait_dscnt 0" ::: "memory");

    // S tiles: s0 -> kv cols 0..15, s1 -> 16..31 (Kt operand = A-layout read)
    v8f s0 = {}, s1 = {};
    {
      const v16h b00 = load_a_frag(&Ks[wave][buf][0][0], CHEAD, lane);
      const v16h b01 = load_a_frag(&Ks[wave][buf][0][32], CHEAD, lane);
      s0 = wmma_f16(qa0, b00, s0);
      s0 = wmma_f16(qa1, b01, s0);
      const v16h b10 = load_a_frag(&Ks[wave][buf][16][0], CHEAD, lane);
      const v16h b11 = load_a_frag(&Ks[wave][buf][16][32], CHEAD, lane);
      s1 = wmma_f16(qa0, b10, s1);
      s1 = wmma_f16(qa1, b11, s1);
    }

    // online softmax (row M = (lane>=16?8:0)+r, cols across the 16-lane half)
#pragma unroll
    for (int r = 0; r < 8; ++r) {
      const float x0 = s0[r] * CSCALE;
      const float x1 = s1[r] * CSCALE;
      float mx = fmaxf(x0, x1);
#pragma unroll
      for (int off = 1; off < 16; off <<= 1) mx = fmaxf(mx, __shfl_xor(mx, off, 32));
      const float mnew = fmaxf(mrow[r], mx);
      const float alpha = __expf(mrow[r] - mnew);
      mrow[r] = mnew;
      const float p0 = __expf(x0 - mnew);
      const float p1 = __expf(x1 - mnew);
      float rs = p0 + p1;
#pragma unroll
      for (int off = 1; off < 16; off <<= 1) rs += __shfl_xor(rs, off, 32);
      lrow[r] = lrow[r] * alpha + rs;
      o0[r] *= alpha; o1[r] *= alpha; o2[r] *= alpha; o3[r] *= alpha;
      const int M = ((lane >> 4) << 3) + r;
      Ps[wave][M][lane & 15]        = (_Float16)p0;
      Ps[wave][M][16 + (lane & 15)] = (_Float16)p1;
    }
    asm volatile("s_wait_dscnt 0" ::: "memory");

    // O += P @ V  (4 column chunks of 16)
    const v16h pa = load_a_frag(&Ps[wave][0][0], 32, lane);
    o0 = wmma_f16(pa, load_b_frag(&Vs[wave][buf][0][0],  CHEAD, lane), o0);
    o1 = wmma_f16(pa, load_b_frag(&Vs[wave][buf][0][16], CHEAD, lane), o1);
    o2 = wmma_f16(pa, load_b_frag(&Vs[wave][buf][0][32], CHEAD, lane), o2);
    o3 = wmma_f16(pa, load_b_frag(&Vs[wave][buf][0][48], CHEAD, lane), o3);
  }

  // normalize and store (f16 intermediate for final projection)
  _Float16* op = Oh + qrow0 * CD + h * CHEAD;
#pragma unroll
  for (int r = 0; r < 8; ++r) {
    const int M = ((lane >> 4) << 3) + r;
    const float inv = 1.0f / lrow[r];
    op[(size_t)M * CD + (lane & 15)]      = (_Float16)(o0[r] * inv);
    op[(size_t)M * CD + 16 + (lane & 15)] = (_Float16)(o1[r] * inv);
    op[(size_t)M * CD + 32 + (lane & 15)] = (_Float16)(o2[r] * inv);
    op[(size_t)M * CD + 48 + (lane & 15)] = (_Float16)(o3[r] * inv);
  }
}

// ---------------------------------------------------------------------------
// Host-side orchestration
// ---------------------------------------------------------------------------
extern "C" void kernel_launch(void* const* d_in, const int* in_sizes, int n_in,
                              void* d_out, int out_size, void* d_ws, size_t ws_size,
                              hipStream_t stream) {
  (void)in_sizes; (void)n_in; (void)out_size; (void)ws_size;
  const float* q    = (const float*)d_in[0];
  const float* k    = (const float*)d_in[1];
  const float* v    = (const float*)d_in[2];
  const float* qpos = (const float*)d_in[3];
  const float* kpos = (const float*)d_in[4];
  const float* gq   = (const float*)d_in[5];
  const float* bq   = (const float*)d_in[6];
  const float* gk   = (const float*)d_in[7];
  const float* bk   = (const float*)d_in[8];
  const float* gv   = (const float*)d_in[9];
  const float* bv   = (const float*)d_in[10];
  const float* Wq   = (const float*)d_in[11];
  const float* Wk   = (const float*)d_in[12];
  const float* Wv   = (const float*)d_in[13];
  const float* Wp   = (const float*)d_in[14];
  const float* bp   = (const float*)d_in[15];

  const size_t BND = (size_t)CB * CNQ * CD;  // 4M elems
  const size_t WSZ = (size_t)CD * CD;        // 1M elems
  _Float16* ws = (_Float16*)d_ws;
  _Float16* Xq  = ws; ws += BND;
  _Float16* Xk  = ws; ws += BND;
  _Float16* Xv  = ws; ws += BND;
  _Float16* Wqh = ws; ws += WSZ;
  _Float16* Wkh = ws; ws += WSZ;
  _Float16* Wvh = ws; ws += WSZ;
  _Float16* Wph = ws; ws += WSZ;
  _Float16* Qh  = ws; ws += BND;
  _Float16* KhB = ws; ws += BND;
  _Float16* VhB = ws; ws += BND;
  _Float16* Ohb = ws; ws += BND;

  // 1) add-pos + LayerNorm -> f16
  ln_kernel<<<CB * CNQ, 256, 0, stream>>>(q, qpos, gq, bq, Xq);
  ln_kernel<<<CB * CNK, 256, 0, stream>>>(k, kpos, gk, bk, Xk);
  ln_kernel<<<CB * CNK, 256, 0, stream>>>(v, nullptr, gv, bv, Xv);

  // 2) weights -> f16
  const int wn = CD * CD;
  cvt_kernel<<<wn / 256, 256, 0, stream>>>(Wq, Wqh, wn);
  cvt_kernel<<<wn / 256, 256, 0, stream>>>(Wk, Wkh, wn);
  cvt_kernel<<<wn / 256, 256, 0, stream>>>(Wv, Wvh, wn);
  cvt_kernel<<<wn / 256, 256, 0, stream>>>(Wp, Wph, wn);

  // 3) projections (block tile 128x128)
  dim3 gg(CD / 128, (CB * CNQ) / 128);
  gemm_kernel<false><<<gg, 256, 0, stream>>>(Xq, Wqh, nullptr, Qh, CD, CD);
  gemm_kernel<false><<<gg, 256, 0, stream>>>(Xk, Wkh, nullptr, KhB, CD, CD);
  gemm_kernel<false><<<gg, 256, 0, stream>>>(Xv, Wvh, nullptr, VhB, CD, CD);

  // 4) attention: B*H*(NQ/16) = 4096 wave-tasks, 2 waves per block
  attn_kernel<<<(CB * CH * (CNQ / 16)) / 2, 64, 0, stream>>>(Qh, KhB, VhB, Ohb);

  // 5) output projection + bias -> f32
  gemm_kernel<true><<<gg, 256, 0, stream>>>(Ohb, Wph, bp, d_out, CD, CD);
}
